// Decoder_10943576670859
// MI455X (gfx1250) — matrix operand order repaired and checked by reading, not compiled, not run on hardware
//
#include <hip/hip_runtime.h>
#include <cstdint>

// ---------------------------------------------------------------- types
typedef unsigned short bf16raw;
typedef __attribute__((ext_vector_type(16))) __bf16 v16bf;
typedef __attribute__((ext_vector_type(8)))  float  v8f;

// ---------------------------------------------------------------- sizes
#define B_SZ   64
#define TIN    512
#define DIN    512
#define MEMD   80
#define STYD   128
#define QD     256
#define DECD   512
#define PRED   256
#define ATTD   128
#define TSTEP  512
#define MR     160            // MEM*R
#define KATT   1152           // (PRE + DIN + STY) + Q   = 896 + 256
#define KDEC   1408           // (Q + DIN + STY) + DEC   = 896 + 512
#define KPROJ  1152           // DEC + DIN + STY

#define CDIV(a,b) (((a)+(b)-1)/(b))

__device__ __forceinline__ bf16raw f2bf(float f) {
  union { float f; unsigned u; } x; x.f = f;
  unsigned u = x.u;
  u += 0x7fffu + ((u >> 16) & 1u);   // round-to-nearest-even
  return (bf16raw)(u >> 16);
}
__device__ __forceinline__ float sigmoidf(float x) { return 1.f / (1.f + __expf(-x)); }

// one b128 async copy global -> LDS (per-lane 16B), tracked by ASYNCcnt
__device__ __forceinline__ void async_b128(unsigned lds_byte_off, const bf16raw* gptr) {
  asm volatile("global_load_async_to_lds_b128 %0, %1, off"
               :: "v"(lds_byte_off), "v"(gptr) : "memory");
}

// ================================================================ GEMM v2
// C[M,N] = A[M,K] * B[N,K]^T (+bias, optional relu). bf16 in, f32 accum.
// Block = 128 threads (4 waves). Block tile = 64(M) x 64(N).
// Double-buffered LDS staging of 64x32 A/B chunks using
// global_load_async_to_lds_b128 + s_wait_asynccnt + barrier; all fragment
// ds_loads are hoisted ahead of the 4 back-to-back v_wmma per wave per chunk.
// Requires: M%64==0, N%64==0, K%32==0.
template <bool BF16OUT>
__global__ void wmma_gemm_lds(const bf16raw* __restrict__ A, int lda,
                              const bf16raw* __restrict__ Bw, int ldb,
                              float* __restrict__ Cf, bf16raw* __restrict__ Cb, int ldc,
                              const float* __restrict__ bias, int relu, int K)
{
  __shared__ bf16raw smA[2][64 * 32];
  __shared__ bf16raw smB[2][64 * 32];
  const int tid  = threadIdx.x;      // 0..127
  const int lane = tid & 31;
  const int w    = tid >> 5;         // wave id = M sub-tile
  const int l15  = lane & 15;
  const int hl   = lane >> 4;
  const int mbase = blockIdx.y * 64;
  const int nbase = blockIdx.x * 64;

  // staging: 64 rows x 32 bf16 = 256 b128 segments per matrix; thread covers
  // segments tid and tid+128 of each. seg j: row j>>2, 16B piece j&3.
  const int r0 = tid >> 2, r1 = r0 + 32, sseg = (tid & 3) * 8; // bf16 elems

  auto stage = [&](int buf, int k0) {
    async_b128((unsigned)(size_t)&smA[buf][r0 * 32 + sseg],
               A + (size_t)(mbase + r0) * lda + k0 + sseg);
    async_b128((unsigned)(size_t)&smA[buf][r1 * 32 + sseg],
               A + (size_t)(mbase + r1) * lda + k0 + sseg);
    async_b128((unsigned)(size_t)&smB[buf][r0 * 32 + sseg],
               Bw + (size_t)(nbase + r0) * ldb + k0 + sseg);
    async_b128((unsigned)(size_t)&smB[buf][r1 * 32 + sseg],
               Bw + (size_t)(nbase + r1) * ldb + k0 + sseg);
  };

  const v8f vzero = {0.f, 0.f, 0.f, 0.f, 0.f, 0.f, 0.f, 0.f};
  v8f acc[4] = {vzero, vzero, vzero, vzero};
  const int akd = hl ? 8 : 0;   // A frag dword offset within 32-elem row
  const int bkd = hl ? 8 : 0;   // B frag dword offset within 32-elem row

  const int nch = K / 32;
  stage(0, 0);
  for (int ci = 0; ci < nch; ++ci) {
    const int buf = ci & 1;
    if (ci + 1 < nch) {
      stage(buf ^ 1, (ci + 1) * 32);
      asm volatile("s_wait_asynccnt 0x4" ::: "memory");   // chunk ci landed
    } else {
      asm volatile("s_wait_asynccnt 0x0" ::: "memory");
    }
    __syncthreads();

    // hoist ALL fragment loads (1 A frag + 4 B frags) before the WMMA quad
    union { v16bf v; unsigned u[8]; } af;
    union { v16bf v; unsigned u[8]; } bb[4];
    const unsigned* aptr = (const unsigned*)&smA[buf][(w * 16 + l15) * 32];
#pragma unroll
    for (int j = 0; j < 4; ++j) {
      af.u[j]     = aptr[akd + j];
      af.u[j + 4] = aptr[akd + 4 + j];
    }
#pragma unroll
    for (int nt = 0; nt < 4; ++nt) {
      const unsigned* bptr = (const unsigned*)&smB[buf][(nt * 16 + l15) * 32];
#pragma unroll
      for (int j = 0; j < 8; ++j) bb[nt].u[j] = bptr[bkd + j];
    }
#pragma unroll
    for (int nt = 0; nt < 4; ++nt)
      acc[nt] = __builtin_amdgcn_wmma_f32_16x16x32_bf16(false, af.v, false, bb[nt].v,
                                                        (short)0, acc[nt], false, false);
    __syncthreads();   // safe to overwrite buf on iteration ci+1's stage
  }

#pragma unroll
  for (int nt = 0; nt < 4; ++nt) {
    const int col = nbase + nt * 16 + l15;
    const float bv = bias ? bias[col] : 0.f;
#pragma unroll
    for (int r = 0; r < 8; ++r) {
      const int row = mbase + w * 16 + r + (hl ? 8 : 0);
      float v = acc[nt][r] + bv;
      if (relu) v = fmaxf(v, 0.f);
      if (BF16OUT) Cb[(size_t)row * ldc + col] = f2bf(v);
      else         Cf[(size_t)row * ldc + col] = v;
    }
  }
}

// ---------------------------------------------------------------- GEMM v1
// one wave per 16x16 tile; used where N%64!=0 (projection, N=160)
template <bool BF16OUT>
__global__ void wmma_gemm(const bf16raw* __restrict__ A, int lda,
                          const bf16raw* __restrict__ Bw, int ldb,
                          float* __restrict__ Cf, bf16raw* __restrict__ Cb, int ldc,
                          const float* __restrict__ bias, int relu, int K)
{
  const int lane = threadIdx.x;
  const int l15  = lane & 15;
  const int hl   = lane >> 4;
  const bf16raw* Arow = A  + (size_t)(blockIdx.y * 16 + l15) * (size_t)lda;
  const bf16raw* Brow = Bw + (size_t)(blockIdx.x * 16 + l15) * (size_t)ldb;
  const int ak = hl ? 16 : 0;
  const int bk = hl ? 16 : 0;
  v8f acc = {0.f, 0.f, 0.f, 0.f, 0.f, 0.f, 0.f, 0.f};
  for (int k0 = 0; k0 < K; k0 += 32) {
    __builtin_prefetch(Arow + k0 + 256, 0, 1);
    __builtin_prefetch(Brow + k0 + 256, 0, 1);
    union { v16bf v; unsigned u[8]; } af, bb;
#pragma unroll
    for (int j = 0; j < 4; ++j) {
      af.u[j]     = *(const unsigned*)(Arow + k0 + ak + 2 * j);
      af.u[j + 4] = *(const unsigned*)(Arow + k0 + ak + 8 + 2 * j);
    }
#pragma unroll
    for (int j = 0; j < 8; ++j)
      bb.u[j] = *(const unsigned*)(Brow + k0 + bk + 2 * j);
    acc = __builtin_amdgcn_wmma_f32_16x16x32_bf16(false, af.v, false, bb.v,
                                                  (short)0, acc, false, false);
  }
  const int col = blockIdx.x * 16 + l15;
  const float bv = bias ? bias[col] : 0.f;
#pragma unroll
  for (int r = 0; r < 8; ++r) {
    const int row = blockIdx.y * 16 + r + (hl ? 8 : 0);
    float v = acc[r] + bv;
    if (relu) v = fmaxf(v, 0.f);
    if (BF16OUT) Cb[(size_t)row * ldc + col] = f2bf(v);
    else         Cf[(size_t)row * ldc + col] = v;
  }
}

// ---------------------------------------------------------------- elementwise
__global__ void f32_to_bf16_k(const float* __restrict__ in, bf16raw* __restrict__ out, int n) {
  int i = blockIdx.x * 256 + threadIdx.x;
  if (i < n) out[i] = f2bf(in[i]);
}

__global__ void add_bias_k(const float* __restrict__ a, const float* __restrict__ b,
                           float* __restrict__ o, int n) {
  int i = blockIdx.x * 256 + threadIdx.x;
  if (i < n) o[i] = a[i] + b[i];
}

__global__ void concat2_bf16(const float* __restrict__ W1, const float* __restrict__ W2,
                             bf16raw* __restrict__ out, int K1, int K2, int total) {
  int i = blockIdx.x * 256 + threadIdx.x;
  if (i >= total) return;
  int K = K1 + K2;
  int n = i / K, k = i % K;
  float v = (k < K1) ? W1[(size_t)n * K1 + k] : W2[(size_t)n * K2 + (k - K1)];
  out[i] = f2bf(v);
}

__global__ void build_prenet_in(const float* __restrict__ memory,
                                const float* __restrict__ init_mem,
                                bf16raw* __restrict__ A0) {
  int i = blockIdx.x * 256 + threadIdx.x;
  if (i >= TSTEP * B_SZ * MR) return;
  int j = i % MR, b = (i / MR) % B_SZ, t = i / (MR * B_SZ);
  float v = (t == 0) ? init_mem[j] : memory[(size_t)b * (TSTEP * MR) + (size_t)(t - 1) * MR + j];
  A0[i] = f2bf(v);
}

__global__ void bcast_row(const float* __restrict__ row, float* __restrict__ out, int Q, int n) {
  int i = blockIdx.x * 256 + threadIdx.x;
  if (i < n) out[i] = row[i % Q];
}
__global__ void zero_f32(float* __restrict__ p, int n) {
  int i = blockIdx.x * 256 + threadIdx.x;
  if (i < n) p[i] = 0.f;
}

__global__ void build_xcat(const bf16raw* __restrict__ pre_t, const float* __restrict__ ctx,
                           const float* __restrict__ style, const float* __restrict__ q,
                           bf16raw* __restrict__ xcat) {
  int i = blockIdx.x * 256 + threadIdx.x;
  if (i >= B_SZ * KATT) return;
  int b = i / KATT, k = i % KATT;
  bf16raw v;
  if      (k < 256)  v = pre_t[b * PRED + k];
  else if (k < 768)  v = f2bf(ctx[b * DIN + (k - 256)]);
  else if (k < 896)  v = f2bf(style[b * STYD + (k - 768)]);
  else               v = f2bf(q[b * QD + (k - 896)]);
  xcat[i] = v;
}

__global__ void build_dxcat(const float* __restrict__ q, const float* __restrict__ ctx,
                            const float* __restrict__ style, const float* __restrict__ dh,
                            bf16raw* __restrict__ dxcat) {
  int i = blockIdx.x * 256 + threadIdx.x;
  if (i >= B_SZ * KDEC) return;
  int b = i / KDEC, k = i % KDEC;
  float v;
  if      (k < 256)  v = q[b * QD + k];
  else if (k < 768)  v = ctx[b * DIN + (k - 256)];
  else if (k < 896)  v = style[b * STYD + (k - 768)];
  else               v = dh[b * DECD + (k - 896)];
  dxcat[i] = f2bf(v);
}

__global__ void build_outcat(const float* __restrict__ dh, const float* __restrict__ ctx,
                             const float* __restrict__ style, bf16raw* __restrict__ oc) {
  int i = blockIdx.x * 256 + threadIdx.x;
  if (i >= B_SZ * KPROJ) return;
  int b = i / KPROJ, k = i % KPROJ;
  float v;
  if      (k < 512)  v = dh[b * DECD + k];
  else if (k < 1024) v = ctx[b * DIN + (k - 512)];
  else               v = style[b * STYD + (k - 1024)];
  oc[i] = f2bf(v);
}

__global__ void lstm_point(const float* __restrict__ z, float* __restrict__ h,
                           float* __restrict__ c, bf16raw* __restrict__ h_bf, int Q) {
  int i = blockIdx.x * 256 + threadIdx.x;
  if (i >= B_SZ * Q) return;
  int b = i / Q, j = i % Q;
  const float* zb = z + (size_t)b * 4 * Q;
  float ig = sigmoidf(zb[j]);
  float fg = sigmoidf(zb[Q + j]);
  float gg = tanhf(zb[2 * Q + j]);
  float og = sigmoidf(zb[3 * Q + j]);
  float cn = fg * c[i] + ig * gg;
  c[i] = cn;
  float hn = og * tanhf(cn);
  h[i] = hn;
  if (h_bf) h_bf[i] = f2bf(hn);
}

__global__ void attn_score(const float* __restrict__ proc, const float* __restrict__ qw,
                           const float* __restrict__ v, const unsigned char* __restrict__ mask,
                           float* __restrict__ e) {
  int idx = blockIdx.x * 256 + threadIdx.x;
  if (idx >= B_SZ * TIN) return;
  int b = idx >> 9;
  const float* p = proc + (size_t)idx * ATTD;
  const float* q = qw + b * ATTD;
  float acc = 0.f;
#pragma unroll 4
  for (int a = 0; a < ATTD; ++a) acc += v[a] * tanhf(p[a] + q[a]);
  e[idx] = mask[idx] ? acc : -1e9f;
}

__global__ void softmax_step(const float* __restrict__ e, float* __restrict__ alpha,
                             float* __restrict__ aligns, int t) {
  int b = blockIdx.x, tid = threadIdx.x;
  __shared__ float red[256];
  const float* eb = e + b * TIN;
  float m = -3e38f;
  for (int i = tid; i < TIN; i += 256) m = fmaxf(m, eb[i]);
  red[tid] = m; __syncthreads();
  for (int s = 128; s > 0; s >>= 1) { if (tid < s) red[tid] = fmaxf(red[tid], red[tid + s]); __syncthreads(); }
  m = red[0]; __syncthreads();
  float sum = 0.f;
  for (int i = tid; i < TIN; i += 256) sum += __expf(eb[i] - m);
  red[tid] = sum; __syncthreads();
  for (int s = 128; s > 0; s >>= 1) { if (tid < s) red[tid] += red[tid + s]; __syncthreads(); }
  float inv = 1.f / red[0];
  float* ab = alpha + b * TIN;
  float* ob = aligns + ((size_t)b * TSTEP + t) * TIN;
  for (int i = tid; i < TIN; i += 256) { float a = __expf(eb[i] - m) * inv; ab[i] = a; ob[i] = a; }
}

__global__ void ctx_kernel(const float* __restrict__ alpha, const float* __restrict__ inputs,
                           float* __restrict__ ctx) {
  int b = blockIdx.x;
  int d = blockIdx.y * 128 + threadIdx.x;
  const float* ab = alpha + b * TIN;
  const float* ib = inputs + (size_t)b * TIN * DIN + d;
  float acc = 0.f;
  for (int t = 0; t < TIN; ++t) acc += ab[t] * ib[(size_t)t * DIN];
  ctx[b * DIN + d] = acc;
}

__global__ void scatter_out(const float* __restrict__ out_t, float* __restrict__ dout, int t) {
  int i = blockIdx.x * 256 + threadIdx.x;
  if (i >= B_SZ * MR) return;
  int b = i / MR, j = i % MR, r = j / MEMD, m = j % MEMD;
  dout[(size_t)b * MEMD * (TSTEP * 2) + (size_t)m * (TSTEP * 2) + 2 * t + r] = out_t[i];
}

// ---------------------------------------------------------------- host
static void glds_f32(const bf16raw* A, int lda, const bf16raw* B, int ldb,
                     float* C, int ldc, const float* bias, int relu,
                     int M, int N, int K, hipStream_t s) {
  wmma_gemm_lds<false><<<dim3(N / 64, M / 64), 128, 0, s>>>(A, lda, B, ldb, C, nullptr, ldc, bias, relu, K);
}
static void glds_bf16(const bf16raw* A, int lda, const bf16raw* B, int ldb,
                      bf16raw* C, int ldc, const float* bias, int relu,
                      int M, int N, int K, hipStream_t s) {
  wmma_gemm_lds<true><<<dim3(N / 64, M / 64), 128, 0, s>>>(A, lda, B, ldb, nullptr, C, ldc, bias, relu, K);
}
static void gsimple_f32(const bf16raw* A, int lda, const bf16raw* B, int ldb,
                        float* C, int ldc, const float* bias, int relu,
                        int M, int N, int K, hipStream_t s) {
  wmma_gemm<false><<<dim3(N / 16, M / 16), 32, 0, s>>>(A, lda, B, ldb, C, nullptr, ldc, bias, relu, K);
}

extern "C" void kernel_launch(void* const* d_in, const int* in_sizes, int n_in,
                              void* d_out, int out_size, void* d_ws, size_t ws_size,
                              hipStream_t stream) {
  (void)in_sizes; (void)n_in; (void)out_size; (void)ws_size;
  const float* inputs    = (const float*)d_in[0];
  const float* memory    = (const float*)d_in[1];
  const float* style     = (const float*)d_in[2];
  const unsigned char* mask = (const unsigned char*)d_in[3];
  const float* prenet_w1 = (const float*)d_in[4];
  const float* prenet_w2 = (const float*)d_in[5];
  const float* attn_Wih  = (const float*)d_in[6];
  const float* attn_Whh  = (const float*)d_in[7];
  const float* attn_bih  = (const float*)d_in[8];
  const float* attn_bhh  = (const float*)d_in[9];
  const float* att_Wq    = (const float*)d_in[10];
  const float* att_Winp  = (const float*)d_in[11];
  const float* att_v     = (const float*)d_in[12];
  const float* dec_Wih   = (const float*)d_in[13];
  const float* dec_Whh   = (const float*)d_in[14];
  const float* dec_bih   = (const float*)d_in[15];
  const float* dec_bhh   = (const float*)d_in[16];
  const float* proj_W    = (const float*)d_in[17];
  const float* proj_b    = (const float*)d_in[18];
  const float* init_query= (const float*)d_in[19];
  const float* init_mem  = (const float*)d_in[20];
  const float* init_dec_h= (const float*)d_in[21];

  float* out_f = (float*)d_out;
  float* aligns = out_f + (size_t)B_SZ * MEMD * (TSTEP * 2);

  char* ws = (char*)d_ws;
  size_t cur = 0;
  auto alloc = [&](size_t bytes) -> char* {
    char* p = ws + cur; cur += (bytes + 255) & ~(size_t)255; return p;
  };
  const size_t MBIG = (size_t)TSTEP * B_SZ;                     // 32768
  bf16raw* inputs_b = (bf16raw*)alloc(MBIG * DIN * 2);
  bf16raw* A0       = (bf16raw*)alloc(MBIG * MR * 2);
  bf16raw* h1       = (bf16raw*)alloc(MBIG * PRED * 2);
  bf16raw* preb     = (bf16raw*)alloc(MBIG * PRED * 2);
  float*   proc     = (float*)  alloc(MBIG * ATTD * 4);
  bf16raw* Wattn    = (bf16raw*)alloc((size_t)4 * QD * KATT * 2);
  bf16raw* Wdec     = (bf16raw*)alloc((size_t)4 * DECD * KDEC * 2);
  bf16raw* Wproj    = (bf16raw*)alloc((size_t)MR * KPROJ * 2);
  bf16raw* Wqb      = (bf16raw*)alloc((size_t)ATTD * QD * 2);
  bf16raw* Winpb    = (bf16raw*)alloc((size_t)ATTD * DIN * 2);
  bf16raw* W1b      = (bf16raw*)alloc((size_t)PRED * MR * 2);
  bf16raw* W2b      = (bf16raw*)alloc((size_t)PRED * PRED * 2);
  float* bias_a = (float*)alloc(4 * QD * 4);
  float* bias_d = (float*)alloc(4 * DECD * 4);
  float* q    = (float*)alloc((size_t)B_SZ * QD * 4);
  float* qc   = (float*)alloc((size_t)B_SZ * QD * 4);
  float* dh   = (float*)alloc((size_t)B_SZ * DECD * 4);
  float* dc   = (float*)alloc((size_t)B_SZ * DECD * 4);
  float* ctx  = (float*)alloc((size_t)B_SZ * DIN * 4);
  bf16raw* qb    = (bf16raw*)alloc((size_t)B_SZ * QD * 2);
  bf16raw* xcat  = (bf16raw*)alloc((size_t)B_SZ * KATT * 2);
  float*   z1    = (float*)  alloc((size_t)B_SZ * 4 * QD * 4);
  float*   qw    = (float*)  alloc((size_t)B_SZ * ATTD * 4);
  float*   ebuf  = (float*)  alloc((size_t)B_SZ * TIN * 4);
  float*   alpha = (float*)  alloc((size_t)B_SZ * TIN * 4);
  bf16raw* dxcat = (bf16raw*)alloc((size_t)B_SZ * KDEC * 2);
  float*   z2    = (float*)  alloc((size_t)B_SZ * 4 * DECD * 4);
  bf16raw* ocat  = (bf16raw*)alloc((size_t)B_SZ * KPROJ * 2);
  float*   out_t = (float*)  alloc((size_t)B_SZ * MR * 4);

  // ---- precompute: conversions / concatenated weights / biases
  int n;
  n = (int)(MBIG * DIN);      f32_to_bf16_k<<<CDIV(n,256),256,0,stream>>>(inputs, inputs_b, n);
  n = ATTD * DIN;             f32_to_bf16_k<<<CDIV(n,256),256,0,stream>>>(att_Winp, Winpb, n);
  n = ATTD * QD;              f32_to_bf16_k<<<CDIV(n,256),256,0,stream>>>(att_Wq, Wqb, n);
  n = PRED * MR;              f32_to_bf16_k<<<CDIV(n,256),256,0,stream>>>(prenet_w1, W1b, n);
  n = PRED * PRED;            f32_to_bf16_k<<<CDIV(n,256),256,0,stream>>>(prenet_w2, W2b, n);
  n = MR * KPROJ;             f32_to_bf16_k<<<CDIV(n,256),256,0,stream>>>(proj_W, Wproj, n);
  n = 4 * QD * KATT;          concat2_bf16<<<CDIV(n,256),256,0,stream>>>(attn_Wih, attn_Whh, Wattn, 896, QD, n);
  n = 4 * DECD * KDEC;        concat2_bf16<<<CDIV(n,256),256,0,stream>>>(dec_Wih, dec_Whh, Wdec, 896, DECD, n);
  n = 4 * QD;                 add_bias_k<<<CDIV(n,256),256,0,stream>>>(attn_bih, attn_bhh, bias_a, n);
  n = 4 * DECD;               add_bias_k<<<CDIV(n,256),256,0,stream>>>(dec_bih, dec_bhh, bias_d, n);
  n = (int)(MBIG * MR);       build_prenet_in<<<CDIV(n,256),256,0,stream>>>(memory, init_mem, A0);

  // ---- precompute GEMMs (parallel, LDS/async double-buffered WMMA)
  glds_bf16(A0, MR, W1b, MR, h1, PRED, nullptr, 1, (int)MBIG, PRED, MR, stream);
  glds_bf16(h1, PRED, W2b, PRED, preb, PRED, nullptr, 1, (int)MBIG, PRED, PRED, stream);
  glds_f32(inputs_b, DIN, Winpb, DIN, proc, ATTD, nullptr, 0, (int)MBIG, ATTD, DIN, stream);

  // ---- initial carry
  n = B_SZ * QD;   bcast_row<<<CDIV(n,256),256,0,stream>>>(init_query, q, QD, n);
  n = B_SZ * QD;   zero_f32<<<CDIV(n,256),256,0,stream>>>(qc, n);
  n = B_SZ * DECD; bcast_row<<<CDIV(n,256),256,0,stream>>>(init_dec_h, dh, DECD, n);
  n = B_SZ * DECD; zero_f32<<<CDIV(n,256),256,0,stream>>>(dc, n);
  n = B_SZ * DIN;  zero_f32<<<CDIV(n,256),256,0,stream>>>(ctx, n);

  // ---- sequential scan over T steps
  for (int t = 0; t < TSTEP; ++t) {
    const bf16raw* pre_t = preb + (size_t)t * B_SZ * PRED;
    n = B_SZ * KATT;
    build_xcat<<<CDIV(n,256),256,0,stream>>>(pre_t, ctx, style, q, xcat);
    glds_f32(xcat, KATT, Wattn, KATT, z1, 4 * QD, bias_a, 0, B_SZ, 4 * QD, KATT, stream);
    n = B_SZ * QD;
    lstm_point<<<CDIV(n,256),256,0,stream>>>(z1, q, qc, qb, QD);
    glds_f32(qb, QD, Wqb, QD, qw, ATTD, nullptr, 0, B_SZ, ATTD, QD, stream);
    n = B_SZ * TIN;
    attn_score<<<CDIV(n,256),256,0,stream>>>(proc, qw, att_v, mask, ebuf);
    softmax_step<<<B_SZ, 256, 0, stream>>>(ebuf, alpha, aligns, t);
    ctx_kernel<<<dim3(B_SZ, DIN / 128), 128, 0, stream>>>(alpha, inputs, ctx);
    n = B_SZ * KDEC;
    build_dxcat<<<CDIV(n,256),256,0,stream>>>(q, ctx, style, dh, dxcat);
    glds_f32(dxcat, KDEC, Wdec, KDEC, z2, 4 * DECD, bias_d, 0, B_SZ, 4 * DECD, KDEC, stream);
    n = B_SZ * DECD;
    lstm_point<<<CDIV(n,256),256,0,stream>>>(z2, dh, dc, nullptr, DECD);
    n = B_SZ * KPROJ;
    build_outcat<<<CDIV(n,256),256,0,stream>>>(dh, ctx, style, ocat);
    gsimple_f32(ocat, KPROJ, Wproj, KPROJ, out_t, MR, proj_b, 0, B_SZ, MR, KPROJ, stream);
    n = B_SZ * MR;
    scatter_out<<<CDIV(n,256),256,0,stream>>>(out_t, out_f, t);
  }
}
